// MultiHeadedAttention_58677843198233
// MI455X (gfx1250) — compile-verified, matrix operand order checked by reading, not run on hardware
//
#include <hip/hip_runtime.h>

// ---------------------------------------------------------------------------
// MI455X (gfx1250) multi-head attention, BF16 WMMA pipeline with
// double-buffered LDS and gfx1250 async global->LDS copies (ASYNCcnt).
// S=4096, E=1024, H=16, D=64.  All matrix math on v_wmma_f32_16x16x32_bf16.
// ---------------------------------------------------------------------------

typedef __attribute__((ext_vector_type(16))) __bf16 v16bf;
typedef __attribute__((ext_vector_type(8)))  float  v8f;

constexpr int SEQ = 4096;
constexpr int EMB = 1024;
constexpr int NH  = 16;
constexpr int HD  = 64;   // head dim

__device__ __forceinline__ __bf16 f2bf(float f) {
    unsigned u = __builtin_bit_cast(unsigned, f);
    unsigned r = u + 0x7FFFu + ((u >> 16) & 1u);   // round-to-nearest-even
    unsigned short hs = (unsigned short)(r >> 16);
    return __builtin_bit_cast(__bf16, hs);
}

union FragBF {
    uint4  q[2];
    v16bf  v;
    __bf16 h[16];
};

// Load a 16-half WMMA fragment whose halves 0..7 live at p[0..7] and
// halves 8..15 at p[16..23] (the ISA 16-bit K-striping). p must be 16B aligned.
__device__ __forceinline__ v16bf ld_frag(const __bf16* p) {
    FragBF f;
    f.q[0] = *(const uint4*)(p);
    f.q[1] = *(const uint4*)(p + 16);
    return f.v;
}

// gfx1250 async global->LDS copy (16 bytes per lane), tracked by ASYNCcnt.
// LDS address = low 32 bits of the generic pointer (flat LDS aperture maps
// addr[31:0] to the LDS offset).
__device__ __forceinline__ void async_copy_b128(const void* lds_dst, const void* gsrc) {
    unsigned lds = (unsigned)(uintptr_t)lds_dst;
    unsigned long long ga = (unsigned long long)(uintptr_t)gsrc;
    asm volatile("global_load_async_to_lds_b128 %0, %1, off"
                 :: "v"(lds), "v"(ga) : "memory");
}
__device__ __forceinline__ void wait_async() {
    asm volatile("s_wait_asynccnt 0" ::: "memory");
}

// ---------------------------------------------------------------------------
// fp32 -> bf16 conversion (vectorized x4)
// ---------------------------------------------------------------------------
__global__ void cvt_bf16x4(const float* __restrict__ src, __bf16* __restrict__ dst, int n4) {
    int i = blockIdx.x * blockDim.x + threadIdx.x;
    if (i < n4) {
        float4 f = ((const float4*)src)[i];
        union { __bf16 h[4]; ushort4 u; } p;
        p.h[0] = f2bf(f.x); p.h[1] = f2bf(f.y); p.h[2] = f2bf(f.z); p.h[3] = f2bf(f.w);
        ((ushort4*)dst)[i] = p.u;
    }
}

// ---------------------------------------------------------------------------
// GEMM: Y[M,N] = X[M,K] (bf16) @ W[K,N] (bf16) + bias (f32)
// Block tile 128x128, BK=32, 8 wave32 waves, double-buffered LDS.
// X tile staged with async global->LDS b128; W tile transposed through VGPRs.
// ---------------------------------------------------------------------------
template <bool OUTF32>
__global__ __launch_bounds__(256) void gemm_bf16(const __bf16* __restrict__ X,
                                                 const __bf16* __restrict__ W,
                                                 const float*  __restrict__ bias,
                                                 void* __restrict__ Yv,
                                                 int M, int N, int K) {
    __shared__ __bf16 As [2][128][40];   // 40-half pitch: conflict-free frag reads
    __shared__ __bf16 BsT[2][128][40];   // W tile stored transposed: BsT[n][k]

    const int tid  = threadIdx.x;
    const int wave = tid >> 5;
    const int lane = tid & 31;
    const int ln   = lane & 15;
    const int lg   = lane >> 4;
    const int wm   = wave & 3;     // 4 row groups of 32
    const int wn   = wave >> 2;    // 2 col groups of 64

    const int blockM = blockIdx.y * 128;
    const int blockN = blockIdx.x * 128;

    auto copy_tiles = [&](int buf, int kb) {
        {   // X tile: 128 rows x 32 halves, pure async copies
            int r = tid >> 1, c = (tid & 1) * 16;
            const __bf16* src = X + (size_t)(blockM + r) * K + kb + c;
            async_copy_b128(&As[buf][r][c],     src);
            async_copy_b128(&As[buf][r][c + 8], src + 8);
        }
        {   // W tile: 32 k-rows x 128 cols, coalesced read, transposed LDS write
            int k  = tid >> 3;
            int nc = (tid & 7) * 16;
            const __bf16* src = W + (size_t)(kb + k) * N + blockN + nc;
            FragBF u;
            u.q[0] = ((const uint4*)src)[0];
            u.q[1] = ((const uint4*)src)[1];
            #pragma unroll
            for (int i = 0; i < 16; ++i) BsT[buf][nc + i][k] = u.h[i];
        }
    };

    v8f acc[2][4];
    #pragma unroll
    for (int mt = 0; mt < 2; ++mt)
        #pragma unroll
        for (int nt = 0; nt < 4; ++nt) acc[mt][nt] = v8f{};

    copy_tiles(0, 0);

    for (int kb = 0; kb < K; kb += 32) {
        const int cur = (kb >> 5) & 1;
        wait_async();          // this wave's async portion of tile t landed
        __syncthreads();       // everyone's tile t (incl. transposed W) visible
        if (kb + 32 < K) copy_tiles(cur ^ 1, kb + 32);   // overlap with compute

        v16bf af[2], bfr[4];
        #pragma unroll
        for (int mt = 0; mt < 2; ++mt)
            af[mt] = ld_frag(&As[cur][wm * 32 + mt * 16 + ln][8 * lg]);
        #pragma unroll
        for (int nt = 0; nt < 4; ++nt)
            bfr[nt] = ld_frag(&BsT[cur][wn * 64 + nt * 16 + ln][8 * lg]);

        #pragma unroll
        for (int mt = 0; mt < 2; ++mt)
            #pragma unroll
            for (int nt = 0; nt < 4; ++nt)
                acc[mt][nt] = __builtin_amdgcn_wmma_f32_16x16x32_bf16(
                    false, af[mt], false, bfr[nt], (short)0, acc[mt][nt], false, false);
    }

    #pragma unroll
    for (int nt = 0; nt < 4; ++nt) {
        const int col = blockN + wn * 64 + nt * 16 + ln;
        const float bv = bias[col];
        #pragma unroll
        for (int mt = 0; mt < 2; ++mt) {
            #pragma unroll
            for (int r = 0; r < 8; ++r) {
                const int row = blockM + wm * 32 + mt * 16 + r + 8 * lg;
                const float v = acc[mt][nt][r] + bv;
                if (OUTF32) ((float*)Yv)[(size_t)row * N + col] = v;
                else        ((__bf16*)Yv)[(size_t)row * N + col] = f2bf(v);
            }
        }
    }
}

// ---------------------------------------------------------------------------
// Fused flash-style attention. One block = one head x 128 query rows.
// 8 waves, each owns a 16-row query strip. K/V chunks of 64 double-buffered
// in LDS; K chunk staged with async global->LDS; V transposed through VGPRs.
// QK^T and P*V both via v_wmma_f32_16x16x32_bf16; online softmax in fp32.
// ---------------------------------------------------------------------------
__global__ __launch_bounds__(256) void attn_kernel(const __bf16* __restrict__ Qg,
                                                   const __bf16* __restrict__ Kg,
                                                   const __bf16* __restrict__ Vg,
                                                   __bf16* __restrict__ Og) {
    __shared__ __bf16 Ks [2][64][72];    // K chunk   [key][d]
    __shared__ __bf16 VsT[2][64][72];    // V chunk transposed [d][key]
    __shared__ __bf16 Ps [8][16][72];    // per-wave P staging [m][key]

    const int tid  = threadIdx.x;
    const int wave = tid >> 5;
    const int lane = tid & 31;
    const int ln   = lane & 15;
    const int lg   = lane >> 4;
    const int head = blockIdx.y;
    const int q0   = blockIdx.x * 128 + wave * 16;

    auto copy_chunk = [&](int buf, int kb) {
        {   // K chunk -> Ks[key][d], pure async copies
            int r = tid >> 2, c = (tid & 3) * 16;
            const __bf16* src = Kg + (size_t)(kb + r) * EMB + head * HD + c;
            async_copy_b128(&Ks[buf][r][c],     src);
            async_copy_b128(&Ks[buf][r][c + 8], src + 8);
        }
        {   // V chunk -> VsT[d][key] (coalesced global read, transposed write)
            int key = tid >> 2, c = (tid & 3) * 16;
            const __bf16* src = Vg + (size_t)(kb + key) * EMB + head * HD + c;
            FragBF u;
            u.q[0] = ((const uint4*)src)[0];
            u.q[1] = ((const uint4*)src)[1];
            #pragma unroll
            for (int i = 0; i < 16; ++i) VsT[buf][c + i][key] = u.h[i];
        }
    };

    // Q fragments for this wave's 16-row strip: 2 K-steps over d=64
    v16bf qf[2];
    #pragma unroll
    for (int ds = 0; ds < 2; ++ds)
        qf[ds] = ld_frag(Qg + (size_t)(q0 + ln) * EMB + head * HD + ds * 32 + 8 * lg);

    v8f oacc[4];
    #pragma unroll
    for (int nt = 0; nt < 4; ++nt) oacc[nt] = v8f{};
    float m8[8], l8[8];
    #pragma unroll
    for (int r = 0; r < 8; ++r) { m8[r] = -1e30f; l8[r] = 0.0f; }

    const float scale = 0.125f;   // 1/sqrt(64)

    copy_chunk(0, 0);

    for (int kb = 0; kb < SEQ; kb += 64) {
        const int cur = (kb >> 6) & 1;
        wait_async();
        __syncthreads();
        if (kb + 64 < SEQ) copy_chunk(cur ^ 1, kb + 64);   // overlap with compute

        // scores: 16 q-rows x 64 keys = 4 N-tiles, K-dim = d = 2 WMMA steps
        v8f sc[4];
        #pragma unroll
        for (int nt = 0; nt < 4; ++nt) sc[nt] = v8f{};
        #pragma unroll
        for (int nt = 0; nt < 4; ++nt)
            #pragma unroll
            for (int ds = 0; ds < 2; ++ds) {
                v16bf kfr = ld_frag(&Ks[cur][nt * 16 + ln][ds * 32 + 8 * lg]);
                sc[nt] = __builtin_amdgcn_wmma_f32_16x16x32_bf16(
                    false, qf[ds], false, kfr, (short)0, sc[nt], false, false);
            }

        // online softmax (rows spread across VGPRs, cols across 16 lanes)
        float pvals[4][8];
        #pragma unroll
        for (int r = 0; r < 8; ++r) {
            float v = -1e30f;
            #pragma unroll
            for (int nt = 0; nt < 4; ++nt) v = fmaxf(v, sc[nt][r]);
            #pragma unroll
            for (int m = 1; m < 16; m <<= 1) v = fmaxf(v, __shfl_xor(v, m, 32));
            v *= scale;
            const float mnew  = fmaxf(m8[r], v);
            const float alpha = __expf(m8[r] - mnew);
            m8[r] = mnew;
            float rs = 0.0f;
            #pragma unroll
            for (int nt = 0; nt < 4; ++nt) {
                float p = __expf(sc[nt][r] * scale - mnew);
                pvals[nt][r] = p;
                rs += p;
            }
            #pragma unroll
            for (int m = 1; m < 16; m <<= 1) rs += __shfl_xor(rs, m, 32);
            l8[r] = l8[r] * alpha + rs;
            #pragma unroll
            for (int nt = 0; nt < 4; ++nt) oacc[nt][r] *= alpha;
        }

        // transpose P (C layout -> A layout) through this wave's LDS staging
        #pragma unroll
        for (int nt = 0; nt < 4; ++nt)
            #pragma unroll
            for (int r = 0; r < 8; ++r)
                Ps[wave][r + 8 * lg][nt * 16 + ln] = f2bf(pvals[nt][r]);

        v16bf pf[2];
        #pragma unroll
        for (int ks = 0; ks < 2; ++ks)
            pf[ks] = ld_frag(&Ps[wave][ln][ks * 32 + 8 * lg]);

        // oacc += P @ V  (K-dim = key = 2 WMMA steps, N-dim = d = 4 tiles)
        #pragma unroll
        for (int nt = 0; nt < 4; ++nt)
            #pragma unroll
            for (int ks = 0; ks < 2; ++ks) {
                v16bf vfr = ld_frag(&VsT[cur][nt * 16 + ln][ks * 32 + 8 * lg]);
                oacc[nt] = __builtin_amdgcn_wmma_f32_16x16x32_bf16(
                    false, pf[ks], false, vfr, (short)0, oacc[nt], false, false);
            }
    }

    #pragma unroll
    for (int r = 0; r < 8; ++r) {
        const float inv = 1.0f / l8[r];
        const int row = q0 + r + 8 * lg;
        #pragma unroll
        for (int nt = 0; nt < 4; ++nt)
            Og[(size_t)row * EMB + head * HD + nt * 16 + ln] = f2bf(oacc[nt][r] * inv);
    }
}

// ---------------------------------------------------------------------------
// Host-side orchestration
// ---------------------------------------------------------------------------
extern "C" void kernel_launch(void* const* d_in, const int* in_sizes, int n_in,
                              void* d_out, int out_size, void* d_ws, size_t ws_size,
                              hipStream_t stream) {
    (void)in_sizes; (void)n_in; (void)out_size; (void)ws_size;

    const float* x  = (const float*)d_in[0];
    const float* Wq = (const float*)d_in[1];
    const float* bq = (const float*)d_in[2];
    const float* Wk = (const float*)d_in[3];
    const float* bk = (const float*)d_in[4];
    const float* Wv = (const float*)d_in[5];
    const float* bv = (const float*)d_in[6];
    const float* Wo = (const float*)d_in[7];
    const float* bo = (const float*)d_in[8];

    char* ws = (char*)d_ws;
    const size_t SZ_X = (size_t)SEQ * EMB * 2;   // 8 MB bf16
    const size_t SZ_W = (size_t)EMB * EMB * 2;   // 2 MB bf16
    __bf16* xb  = (__bf16*)(ws);
    __bf16* Wqb = (__bf16*)(ws + SZ_X);
    __bf16* Wkb = (__bf16*)(ws + SZ_X + SZ_W);
    __bf16* Wvb = (__bf16*)(ws + SZ_X + 2 * SZ_W);
    __bf16* Wob = (__bf16*)(ws + SZ_X + 3 * SZ_W);
    __bf16* Qb  = (__bf16*)(ws + SZ_X + 4 * SZ_W);
    __bf16* Kb  = (__bf16*)(ws + 2 * SZ_X + 4 * SZ_W);
    __bf16* Vb  = (__bf16*)(ws + 3 * SZ_X + 4 * SZ_W);
    __bf16* Ab  = (__bf16*)(ws + 4 * SZ_X + 4 * SZ_W);

    // 1) fp32 -> bf16 conversions
    const int nx4 = SEQ * EMB / 4;
    const int nw4 = EMB * EMB / 4;
    cvt_bf16x4<<<nx4 / 256, 256, 0, stream>>>(x,  xb,  nx4);
    cvt_bf16x4<<<nw4 / 256, 256, 0, stream>>>(Wq, Wqb, nw4);
    cvt_bf16x4<<<nw4 / 256, 256, 0, stream>>>(Wk, Wkb, nw4);
    cvt_bf16x4<<<nw4 / 256, 256, 0, stream>>>(Wv, Wvb, nw4);
    cvt_bf16x4<<<nw4 / 256, 256, 0, stream>>>(Wo, Wob, nw4);

    // 2) Q/K/V projections (bf16 out)
    dim3 gg(EMB / 128, SEQ / 128);
    gemm_bf16<false><<<gg, 256, 0, stream>>>(xb, Wqb, bq, Qb, SEQ, EMB, EMB);
    gemm_bf16<false><<<gg, 256, 0, stream>>>(xb, Wkb, bk, Kb, SEQ, EMB, EMB);
    gemm_bf16<false><<<gg, 256, 0, stream>>>(xb, Wvb, bv, Vb, SEQ, EMB, EMB);

    // 3) fused attention
    attn_kernel<<<dim3(SEQ / 128, NH), 256, 0, stream>>>(Qb, Kb, Vb, Ab);

    // 4) output projection (fp32 out)
    gemm_bf16<true><<<gg, 256, 0, stream>>>(Ab, Wob, bo, (float*)d_out, SEQ, EMB, EMB);
}
